// Model_25022479466544
// MI455X (gfx1250) — compile-verified, matrix operand order checked by reading, not run on hardware
//
#include <hip/hip_runtime.h>

// ---------------------------------------------------------------------------
// Model: regroup GEMM -> 2x BiLSTM -> attention -> post -> intents
// B=32, T=512, F=2048, H=768, D=1024, H2=1536, NI=60
// bf16 WMMA (v_wmma_f32_16x16x32_bf16) for all large GEMMs, with
// double-buffered async global->LDS staging (ASYNCcnt) in the GEMM;
// per-timestep fused recurrent kernel with gate-interleaved weight packing.
// ---------------------------------------------------------------------------

#define B_  32
#define T_  512
#define F_  2048
#define H_  768
#define D_  1024
#define NI_ 60
#define H2_ 1536
#define G4_ 3072   // 4*H (packed gate columns)

typedef __attribute__((ext_vector_type(16))) __bf16 v16bf;
typedef __attribute__((ext_vector_type(8)))  float  v8f;

union FragU { uint4 q[2]; v16bf v; };

__device__ __forceinline__ unsigned short f2bf(float x) {
  unsigned u = __float_as_uint(x);
  unsigned r = u + 0x7FFFu + ((u >> 16) & 1u);   // round-to-nearest-even
  if ((u & 0x7F800000u) == 0x7F800000u) r = u;   // inf/nan passthrough
  return (unsigned short)(r >> 16);
}
__device__ __forceinline__ float bf2f(unsigned short h) {
  return __uint_as_float(((unsigned)h) << 16);
}
__device__ __forceinline__ float sigf(float x) { return 1.0f / (1.0f + __expf(-x)); }

__device__ __forceinline__ v8f zero8() {
  v8f z;
#pragma unroll
  for (int r = 0; r < 8; ++r) z[r] = 0.0f;
  return z;
}

// A-fragment (16x32 bf16) loader, §7.12.2 layout:
// lane L holds row (L&15); lanes 0-15 hold K chunks [0..7],[16..23],
// lanes 16-31 hold [8..15],[24..31].  B fragments use the same pattern
// on a pre-transposed [N,K] matrix.
__device__ __forceinline__ v16bf load_frag16(const unsigned short* __restrict__ p0,
                                             int ld, int lane) {
  const int r  = lane & 15;
  const int ks = lane >> 4;      // 0 or 1
  const unsigned short* p = p0 + (size_t)r * ld + ks * 8;
  FragU f;
  f.q[0] = *reinterpret_cast<const uint4*>(p);
  f.q[1] = *reinterpret_cast<const uint4*>(p + 16);
  return f.v;
}

__device__ __forceinline__ v8f wmma_bf16(v16bf a, v16bf b, v8f c) {
  return __builtin_amdgcn_wmma_f32_16x16x32_bf16(false, a, false, b, (short)0, c,
                                                 false, false);
}

// Async global->LDS copy of one 16B chunk per lane (tracked by ASYNCcnt).
__device__ __forceinline__ void async_ld16(unsigned lds_off, const void* gptr) {
  asm volatile("global_load_async_to_lds_b128 %0, %1, off"
               :: "v"(lds_off), "v"(gptr) : "memory");
}
__device__ __forceinline__ void wait_async4() {
  asm volatile("s_wait_asynccnt 0x4" ::: "memory");
}
__device__ __forceinline__ void wait_async0() {
  asm volatile("s_wait_asynccnt 0x0" ::: "memory");
}
__device__ __forceinline__ unsigned lds_off32(const void* p) {
  return (unsigned)(unsigned long long)p;   // low 32 bits = LDS offset
}

// Straight-line async stage issue for the GEMM: each thread copies 4x16B
// (A rows row0/row0+64 and B rows row0/row0+64 at column chunk kc0).
struct AsyncStage {
  const unsigned short *gA0, *gA1, *gB0, *gB1;  // advance 64B (32 elems)/stage
  unsigned lA, lB;                               // LDS offsets in buffer 0
  __device__ __forceinline__ void issue(int buf) {
    const unsigned o = buf ? 8192u : 0u;         // buffer stride: 128*32*2B
    async_ld16(lA + o,         gA0);
    async_ld16(lA + o + 4096u, gA1);             // +64 rows * 32 cols * 2B
    async_ld16(lB + o,         gB0);
    async_ld16(lB + o + 4096u, gB1);
    gA0 += 32; gA1 += 32; gB0 += 32; gB1 += 32;
  }
};

// ---------------------------------------------------------------------------
// Tiled WMMA GEMM: C[M,N] = A[M,K] * Bt[N,K]^T (+bias).
// Block 256 threads = 8 waves (2 M-waves x 4 N-waves), tile 128x128x32.
// Full-M tiles: async double-buffered global->LDS staging (4 ops/wave/stage).
// M-tail tile (only the M=32 post-GEMM): guarded synchronous path.
// Requires: K % 32 == 0, N % 128 == 0.
// out_bf: 0 -> fp32 C, 1 -> bf16 C (stored as ushort).
// ---------------------------------------------------------------------------
__global__ __launch_bounds__(256) void gemm_bf16_wmma(
    const unsigned short* __restrict__ A,
    const unsigned short* __restrict__ Bt,
    const float* __restrict__ bias,
    float* __restrict__ Cf,
    unsigned short* __restrict__ Cb,
    int M, int N, int K, int out_bf)
{
  __shared__ __align__(16) unsigned short sA[2][128 * 32];
  __shared__ __align__(16) unsigned short sB[2][128 * 32];
  const int tid  = threadIdx.x;
  const int lane = tid & 31;
  const int wave = tid >> 5;
  const int wm = wave & 1;        // 0..1  (64 rows each)
  const int wn = wave >> 1;       // 0..3  (32 cols each)
  const int bm0 = blockIdx.y * 128;
  const int bn0 = blockIdx.x * 128;
  const int nk  = K >> 5;

  v8f acc[4][2];
#pragma unroll
  for (int i = 0; i < 4; ++i) { acc[i][0] = zero8(); acc[i][1] = zero8(); }

  if (bm0 + 128 <= M) {
    // ---- fast path: async double-buffered staging, straight-line issue ----
    const int row0 = tid >> 2;             // 0..63
    const int kc0  = (tid & 3) << 3;       // 0,8,16,24
    AsyncStage st;
    st.gA0 = A  + (size_t)(bm0 + row0) * K + kc0;
    st.gA1 = st.gA0 + (size_t)64 * K;
    st.gB0 = Bt + (size_t)(bn0 + row0) * K + kc0;
    st.gB1 = st.gB0 + (size_t)64 * K;
    st.lA  = lds_off32(&sA[0][row0 * 32 + kc0]);
    st.lB  = lds_off32(&sB[0][row0 * 32 + kc0]);

    st.issue(0);                           // prologue stage
    for (int it = 0; it < nk; ++it) {
      const int cur = it & 1;
      if (it + 1 < nk) {
        st.issue(cur ^ 1);
        wait_async4();                     // previous stage has landed
      } else {
        wait_async0();
      }
      __syncthreads();

      v16bf bf0 = load_frag16(sB[cur] + (wn * 32 + 0)  * 32, 32, lane);
      v16bf bf1 = load_frag16(sB[cur] + (wn * 32 + 16) * 32, 32, lane);
#pragma unroll
      for (int mi = 0; mi < 4; ++mi) {
        v16bf af = load_frag16(sA[cur] + (wm * 64 + mi * 16) * 32, 32, lane);
        acc[mi][0] = wmma_bf16(af, bf0, acc[mi][0]);
        acc[mi][1] = wmma_bf16(af, bf1, acc[mi][1]);
      }
      __syncthreads();   // all waves done reading buf[cur] before it is refilled
    }
  } else {
    // ---- guarded synchronous path (M tail) ----
    for (int k0 = 0; k0 < K; k0 += 32) {
      for (int c = tid; c < 512; c += 256) {
        int row = c >> 2, kc = (c & 3) << 3;
        uint4 val;
        int gm = bm0 + row;
        if (gm < M) val = *reinterpret_cast<const uint4*>(A + (size_t)gm * K + k0 + kc);
        else { val.x = val.y = val.z = val.w = 0u; }
        *reinterpret_cast<uint4*>(&sA[0][row * 32 + kc]) = val;
        *reinterpret_cast<uint4*>(&sB[0][row * 32 + kc]) =
            *reinterpret_cast<const uint4*>(Bt + (size_t)(bn0 + row) * K + k0 + kc);
      }
      __syncthreads();
      v16bf bf0 = load_frag16(sB[0] + (wn * 32 + 0)  * 32, 32, lane);
      v16bf bf1 = load_frag16(sB[0] + (wn * 32 + 16) * 32, 32, lane);
#pragma unroll
      for (int mi = 0; mi < 4; ++mi) {
        v16bf af = load_frag16(sA[0] + (wm * 64 + mi * 16) * 32, 32, lane);
        acc[mi][0] = wmma_bf16(af, bf0, acc[mi][0]);
        acc[mi][1] = wmma_bf16(af, bf1, acc[mi][1]);
      }
      __syncthreads();
    }
  }

  // C/D layout: VGPR r -> rows r (lanes 0-15) and 8+r (lanes 16-31), col = lane&15
  const int cm = (lane < 16) ? 0 : 8;
  const int cn = lane & 15;
#pragma unroll
  for (int mi = 0; mi < 4; ++mi) {
#pragma unroll
    for (int ni = 0; ni < 2; ++ni) {
      int gn = bn0 + wn * 32 + ni * 16 + cn;
      float bv = bias ? bias[gn] : 0.0f;
#pragma unroll
      for (int r = 0; r < 8; ++r) {
        int gm = bm0 + wm * 64 + mi * 16 + cm + r;
        if (gm < M) {
          float v = acc[mi][ni][r] + bv;
          if (out_bf) Cb[(size_t)gm * N + gn] = f2bf(v);
          else        Cf[(size_t)gm * N + gn] = v;
        }
      }
    }
  }
}

// ---------------------------------------------------------------------------
// One LSTM timestep, gate-packed (col p=4j+g), fused GEMM + gate update.
// M = B_ = 32 rows (2 M-frags), N = 3072 packed gate cols, K = 768.
// Grid: 24 blocks x 256 threads.  LDS: h tile (48KB, async-filled),
// reused as z tile after the K loop.
// ---------------------------------------------------------------------------
__global__ __launch_bounds__(256) void lstm_step(
    const float* __restrict__ Xp,            // [B_*T, 3072] (row = b*T + t), bias folded in
    const unsigned short* __restrict__ WhT,  // [3072, 768] bf16, gate-packed, transposed
    const unsigned short* __restrict__ h_in, // [32, 768] bf16
    unsigned short* __restrict__ h_out,      // [32, 768] bf16
    float* __restrict__ cst,                 // [32, 768] fp32 cell state
    unsigned short* __restrict__ hs,         // [B_*T, 768] bf16 output sequence
    int t, int T)
{
  __shared__ __align__(16) unsigned short smem[B_ * H_];   // 48KB, reused as z
  const int tid  = threadIdx.x;
  const int lane = tid & 31;
  const int wave = tid >> 5;

  // warm L2 for next timestep's Xp slice of this block (256 x 64B lines = 16KB)
  if (t + 1 < T) {
    const float* pf = Xp + ((size_t)(tid & 31) * T + (t + 1)) * G4_ +
                      blockIdx.x * 128 + (tid >> 5) * 16;
    __builtin_prefetch(pf, 0, 1);
  }

  // async-fill the 48KB h tile: 3072 x 16B chunks, 12 per thread, straight-line
  {
    const unsigned      l0 = lds_off32(smem) + (unsigned)tid * 16u;
    const unsigned short* g = h_in + tid * 8;
#pragma unroll
    for (int ch = 0; ch < 12; ++ch)
      async_ld16(l0 + ch * 4096u, g + ch * 2048);
    wait_async0();
  }
  __syncthreads();

  const int n0 = blockIdx.x * 128 + wave * 16;
  v8f a0 = zero8(), a1 = zero8();
#pragma unroll 4
  for (int k0 = 0; k0 < H_; k0 += 32) {
    v16bf bf  = load_frag16(WhT + (size_t)n0 * H_ + k0, H_, lane);
    v16bf af0 = load_frag16(smem + 0  * H_ + k0, H_, lane);
    v16bf af1 = load_frag16(smem + 16 * H_ + k0, H_, lane);
    a0 = wmma_bf16(af0, bf, a0);
    a1 = wmma_bf16(af1, bf, a1);
  }
  __syncthreads();                      // h tile fully consumed

  float* z = reinterpret_cast<float*>(smem);   // [32][128]
  const int cm = (lane < 16) ? 0 : 8;
  const int cn = lane & 15;
  const int lcol = wave * 16 + cn;
  const int gcol = blockIdx.x * 128 + lcol;
#pragma unroll
  for (int r = 0; r < 8; ++r) {
    int row0 = r + cm;
    z[row0 * 128 + lcol] = a0[r] + Xp[((size_t)row0 * T + t) * G4_ + gcol];
    int row1 = 16 + r + cm;
    z[row1 * 128 + lcol] = a1[r] + Xp[((size_t)row1 * T + t) * G4_ + gcol];
  }
  __syncthreads();

  for (int i = tid; i < B_ * 32; i += 256) {   // 32 rows x 32 j's per block
    int row = i & 31;
    int jj  = i >> 5;                          // 0..31
    float zi = z[row * 128 + jj * 4 + 0];
    float zf = z[row * 128 + jj * 4 + 1];
    float zg = z[row * 128 + jj * 4 + 2];
    float zo = z[row * 128 + jj * 4 + 3];
    int j = blockIdx.x * 32 + jj;
    size_t ci = (size_t)row * H_ + j;
    float cc = sigf(zf) * cst[ci] + sigf(zi) * tanhf(zg);
    cst[ci] = cc;
    float hh = sigf(zo) * tanhf(cc);
    unsigned short hb = f2bf(hh);
    h_out[ci] = hb;
    hs[((size_t)row * T + t) * H_ + j] = hb;
  }
}

// ---------------------------------------------------------------------------
// Small helper kernels
// ---------------------------------------------------------------------------
__global__ void cast_f32_bf16(const float* __restrict__ in,
                              unsigned short* __restrict__ out, size_t n) {
  size_t i = (size_t)blockIdx.x * blockDim.x + threadIdx.x;
  if (i < n) out[i] = f2bf(in[i]);
}

__global__ void zero_u32(unsigned* __restrict__ p, int n) {
  int i = blockIdx.x * blockDim.x + threadIdx.x;
  if (i < n) p[i] = 0u;
}

// out[n*K+k] = (bf16) in[k*N+n]
__global__ void transpose_cast(const float* __restrict__ in,
                               unsigned short* __restrict__ out, int K, int N) {
  size_t i = (size_t)blockIdx.x * blockDim.x + threadIdx.x;
  if (i >= (size_t)K * N) return;
  int n = (int)(i / K), k = (int)(i % K);
  out[i] = f2bf(in[(size_t)k * N + n]);
}

// Gate-interleaved transpose: packed row p = 4j+g <- original column g*H + j
__global__ void pack_gate_T(const float* __restrict__ in,
                            unsigned short* __restrict__ out, int K, int H) {
  size_t i = (size_t)blockIdx.x * blockDim.x + threadIdx.x;
  if (i >= (size_t)4 * H * K) return;
  int p = (int)(i / K), k = (int)(i % K);
  int g = p & 3, j = p >> 2;
  out[i] = f2bf(in[(size_t)k * 4 * H + (size_t)g * H + j]);
}

__global__ void pack_bias(const float* __restrict__ b, float* __restrict__ bp, int H) {
  int p = blockIdx.x * blockDim.x + threadIdx.x;
  if (p >= 4 * H) return;
  bp[p] = b[(p & 3) * H + (p >> 2)];
}

// flip within valid length: out[b,t,:] = in[b,(T-1-t+len[b])%T,:]  (bf16, vec8)
__global__ void flip_bf16(const unsigned short* __restrict__ in,
                          unsigned short* __restrict__ out,
                          const int* __restrict__ lengths, int T, int Fdim) {
  int F8 = Fdim >> 3;
  size_t nchunk = (size_t)B_ * T * F8;
  size_t i = (size_t)blockIdx.x * blockDim.x + threadIdx.x;
  if (i >= nchunk) return;
  int f8 = (int)(i % F8);
  size_t r = i / F8;
  int t = (int)(r % T), b = (int)(r / T);
  int src = (T - 1 - t + lengths[b]) % T;
  *reinterpret_cast<uint4*>(out + ((size_t)(b * T + t) * Fdim) + f8 * 8) =
      *reinterpret_cast<const uint4*>(in + ((size_t)(b * T + src) * Fdim) + f8 * 8);
}

// x_cat[b,t,0:768]=hsF[b,t]; x_cat[b,t,768:1536]=hsB[b,flip(t)]   (bf16, vec8)
__global__ void concat_unflip(const unsigned short* __restrict__ hsF,
                              const unsigned short* __restrict__ hsB,
                              const int* __restrict__ lengths,
                              unsigned short* __restrict__ out, int T) {
  size_t i = (size_t)blockIdx.x * blockDim.x + threadIdx.x;
  size_t nchunk = (size_t)B_ * T * (H2_ / 8);
  if (i >= nchunk) return;
  int ch = (int)(i % (H2_ / 8));
  size_t r = i / (H2_ / 8);
  int t = (int)(r % T), b = (int)(r / T);
  uint4 v;
  if (ch < H_ / 8) {
    v = *reinterpret_cast<const uint4*>(hsF + (size_t)(b * T + t) * H_ + ch * 8);
  } else {
    int src = (T - 1 - t + lengths[b]) % T;
    v = *reinterpret_cast<const uint4*>(hsB + (size_t)(b * T + src) * H_ + (ch - H_ / 8) * 8);
  }
  *reinterpret_cast<uint4*>(out + (size_t)(b * T + t) * H2_ + ch * 8) = v;
}

// scores[row] = sum_d tanh(keys[row,d]) * we[d]  (masked by length)
__global__ __launch_bounds__(256) void scores_kernel(
    const float* __restrict__ keys, const float* __restrict__ we,
    const int* __restrict__ lengths, float* __restrict__ scores, int T) {
  __shared__ float red[256];
  int row = blockIdx.x, tid = threadIdx.x;
  float s = 0.0f;
  for (int d = tid; d < H2_; d += 256)
    s += tanhf(keys[(size_t)row * H2_ + d]) * we[d];
  red[tid] = s; __syncthreads();
  for (int st = 128; st > 0; st >>= 1) {
    if (tid < st) red[tid] += red[tid + st];
    __syncthreads();
  }
  if (tid == 0) {
    int b = row / T, t = row % T;
    scores[row] = (t < lengths[b]) ? red[0] : -__builtin_huge_valf();
  }
}

__global__ __launch_bounds__(256) void softmax_kernel(
    const float* __restrict__ sc, float* __restrict__ attn, int T) {
  __shared__ float red[256];
  int b = blockIdx.x, tid = threadIdx.x;
  float m = -__builtin_huge_valf();
  for (int t = tid; t < T; t += 256) m = fmaxf(m, sc[b * T + t]);
  red[tid] = m; __syncthreads();
  for (int st = 128; st > 0; st >>= 1) {
    if (tid < st) red[tid] = fmaxf(red[tid], red[tid + st]);
    __syncthreads();
  }
  float mx = red[0]; __syncthreads();
  float sum = 0.0f;
  for (int t = tid; t < T; t += 256) {
    float e = __expf(sc[b * T + t] - mx);
    attn[b * T + t] = e;
    sum += e;
  }
  red[tid] = sum; __syncthreads();
  for (int st = 128; st > 0; st >>= 1) {
    if (tid < st) red[tid] += red[tid + st];
    __syncthreads();
  }
  float inv = 1.0f / red[0];
  for (int t = tid; t < T; t += 256) attn[b * T + t] *= inv;
}

// ctx[b,d] = sum_t attn[b,t] * x2[b,t,d]   (bf16 out for the post GEMM)
__global__ void context_kernel(const float* __restrict__ attn,
                               const unsigned short* __restrict__ x2,
                               unsigned short* __restrict__ ctx, int T) {
  int b = blockIdx.x;
  int d = blockIdx.y * 256 + threadIdx.x;
  float s = 0.0f;
  for (int t = 0; t < T; ++t)
    s += attn[b * T + t] * bf2f(x2[(size_t)(b * T + t) * H2_ + d]);
  ctx[b * H2_ + d] = f2bf(s);
}

__global__ void intents_kernel(const float* __restrict__ ctxP,
                               const float* __restrict__ W_int,
                               const float* __restrict__ b_int,
                               float* __restrict__ out) {
  int i = blockIdx.x * 256 + threadIdx.x;
  if (i >= B_ * NI_) return;
  int b = i / NI_, k = i % NI_;
  float s = b_int[k];
  for (int d = 0; d < H2_; ++d) s += ctxP[b * H2_ + d] * W_int[d * NI_ + k];
  out[i] = s;
}

// ---------------------------------------------------------------------------
// Launcher
// ---------------------------------------------------------------------------
static inline unsigned gb(size_t n) { return (unsigned)((n + 255) / 256); }

extern "C" void kernel_launch(void* const* d_in, const int* in_sizes, int n_in,
                              void* d_out, int out_size, void* d_ws, size_t ws_size,
                              hipStream_t stream) {
  (void)in_sizes; (void)n_in; (void)out_size; (void)ws_size;

  const float* lat       = (const float*)d_in[0];
  const int*   lengths   = (const int*)d_in[1];
  const float* W_regroup = (const float*)d_in[2];
  const float* b_regroup = (const float*)d_in[3];
  const float* Wi1f = (const float*)d_in[4];
  const float* Wh1f = (const float*)d_in[5];
  const float* b1f  = (const float*)d_in[6];
  const float* Wi1b = (const float*)d_in[7];
  const float* Wh1b = (const float*)d_in[8];
  const float* b1b  = (const float*)d_in[9];
  const float* Wi2f = (const float*)d_in[10];
  const float* Wh2f = (const float*)d_in[11];
  const float* b2f  = (const float*)d_in[12];
  const float* Wi2b = (const float*)d_in[13];
  const float* Wh2b = (const float*)d_in[14];
  const float* b2b  = (const float*)d_in[15];
  const float* W_keys   = (const float*)d_in[16];
  const float* w_energy = (const float*)d_in[17];
  const float* W_post   = (const float*)d_in[18];
  const float* b_post   = (const float*)d_in[19];
  const float* W_int    = (const float*)d_in[20];
  const float* b_int    = (const float*)d_in[21];
  float* out = (float*)d_out;

  // ---- workspace bump allocator (~770 MB total) ----
  char* ws = (char*)d_ws;
  size_t off = 0;
  auto alloc = [&](size_t bytes) -> char* {
    off = (off + 255) & ~(size_t)255;
    char* p = ws + off;
    off += bytes;
    return p;
  };
  const size_t MT = (size_t)B_ * T_;   // 16384

  unsigned short* latB  = (unsigned short*)alloc(MT * F_ * 2);
  unsigned short* WrT   = (unsigned short*)alloc((size_t)D_ * F_ * 2);
  unsigned short* Wi1fP = (unsigned short*)alloc((size_t)G4_ * D_ * 2);
  unsigned short* Wi1bP = (unsigned short*)alloc((size_t)G4_ * D_ * 2);
  unsigned short* Wh1fP = (unsigned short*)alloc((size_t)G4_ * H_ * 2);
  unsigned short* Wh1bP = (unsigned short*)alloc((size_t)G4_ * H_ * 2);
  unsigned short* Wi2fP = (unsigned short*)alloc((size_t)G4_ * H2_ * 2);
  unsigned short* Wi2bP = (unsigned short*)alloc((size_t)G4_ * H2_ * 2);
  unsigned short* Wh2fP = (unsigned short*)alloc((size_t)G4_ * H_ * 2);
  unsigned short* Wh2bP = (unsigned short*)alloc((size_t)G4_ * H_ * 2);
  float* b1fP = (float*)alloc(G4_ * 4);
  float* b1bP = (float*)alloc(G4_ * 4);
  float* b2fP = (float*)alloc(G4_ * 4);
  float* b2bP = (float*)alloc(G4_ * 4);
  unsigned short* WkT = (unsigned short*)alloc((size_t)H2_ * H2_ * 2);
  unsigned short* WpT = (unsigned short*)alloc((size_t)H2_ * H2_ * 2);

  unsigned short* xB     = (unsigned short*)alloc(MT * D_ * 2);
  unsigned short* xFlipB = (unsigned short*)alloc(MT * H2_ * 2);  // sized for layer2
  unsigned short* x1B    = (unsigned short*)alloc(MT * H2_ * 2);
  unsigned short* x2B    = (unsigned short*)alloc(MT * H2_ * 2);
  float* XpF = (float*)alloc(MT * G4_ * 4);
  float* XpB = (float*)alloc(MT * G4_ * 4);
  unsigned short* hsF = (unsigned short*)alloc(MT * H_ * 2);
  unsigned short* hsB = (unsigned short*)alloc(MT * H_ * 2);
  unsigned short* hA  = (unsigned short*)alloc((size_t)B_ * H_ * 2);
  unsigned short* hBf = (unsigned short*)alloc((size_t)B_ * H_ * 2);
  float* cBuf   = (float*)alloc((size_t)B_ * H_ * 4);
  float* scores = (float*)alloc(MT * 4);
  float* attn   = (float*)alloc(MT * 4);
  unsigned short* ctxB = (unsigned short*)alloc((size_t)B_ * H2_ * 2);
  float* ctxP = (float*)alloc((size_t)B_ * H2_ * 4);
  float* keys = XpF;   // alias: Xp buffers are dead by attention time

  // ---- phase 0: cast input + pack weights (bf16, gate-interleaved, [N,K]) ----
  cast_f32_bf16<<<gb(MT * F_), 256, 0, stream>>>(lat, latB, MT * F_);
  transpose_cast<<<gb((size_t)F_ * D_), 256, 0, stream>>>(W_regroup, WrT, F_, D_);
  pack_gate_T<<<gb((size_t)G4_ * D_), 256, 0, stream>>>(Wi1f, Wi1fP, D_, H_);
  pack_gate_T<<<gb((size_t)G4_ * D_), 256, 0, stream>>>(Wi1b, Wi1bP, D_, H_);
  pack_gate_T<<<gb((size_t)G4_ * H_), 256, 0, stream>>>(Wh1f, Wh1fP, H_, H_);
  pack_gate_T<<<gb((size_t)G4_ * H_), 256, 0, stream>>>(Wh1b, Wh1bP, H_, H_);
  pack_gate_T<<<gb((size_t)G4_ * H2_), 256, 0, stream>>>(Wi2f, Wi2fP, H2_, H_);
  pack_gate_T<<<gb((size_t)G4_ * H2_), 256, 0, stream>>>(Wi2b, Wi2bP, H2_, H_);
  pack_gate_T<<<gb((size_t)G4_ * H_), 256, 0, stream>>>(Wh2f, Wh2fP, H_, H_);
  pack_gate_T<<<gb((size_t)G4_ * H_), 256, 0, stream>>>(Wh2b, Wh2bP, H_, H_);
  pack_bias<<<gb(G4_), 256, 0, stream>>>(b1f, b1fP, H_);
  pack_bias<<<gb(G4_), 256, 0, stream>>>(b1b, b1bP, H_);
  pack_bias<<<gb(G4_), 256, 0, stream>>>(b2f, b2fP, H_);
  pack_bias<<<gb(G4_), 256, 0, stream>>>(b2b, b2bP, H_);
  transpose_cast<<<gb((size_t)H2_ * H2_), 256, 0, stream>>>(W_keys, WkT, H2_, H2_);
  transpose_cast<<<gb((size_t)H2_ * H2_), 256, 0, stream>>>(W_post, WpT, H2_, H2_);

  // ---- phase 1: regroup GEMM -> x (bf16) ----
  {
    dim3 g(D_ / 128, (unsigned)(MT / 128));
    gemm_bf16_wmma<<<g, 256, 0, stream>>>(latB, WrT, b_regroup, nullptr, xB,
                                          (int)MT, D_, F_, 1);
  }

  // per-direction recurrence driver
  auto run_dir = [&](const float* Xp, const unsigned short* WhT, unsigned short* hs) {
    zero_u32<<<gb(B_ * H_), 256, 0, stream>>>((unsigned*)cBuf, B_ * H_);
    zero_u32<<<gb(B_ * H_ / 2), 256, 0, stream>>>((unsigned*)hA, B_ * H_ / 2);
    const unsigned short* hin = hA;
    unsigned short* hout = hBf;
    for (int t = 0; t < T_; ++t) {
      lstm_step<<<G4_ / 128, 256, 0, stream>>>(Xp, WhT, hin, hout, cBuf, hs, t, T_);
      unsigned short* tmp = (unsigned short*)hin;
      hin = hout;
      hout = tmp;
    }
  };

  // ---- phase 2: BiLSTM layer 1 ----
  flip_bf16<<<gb(MT * (D_ / 8)), 256, 0, stream>>>(xB, xFlipB, lengths, T_, D_);
  {
    dim3 g(G4_ / 128, (unsigned)(MT / 128));
    gemm_bf16_wmma<<<g, 256, 0, stream>>>(xB,     Wi1fP, b1fP, XpF, nullptr, (int)MT, G4_, D_, 0);
    gemm_bf16_wmma<<<g, 256, 0, stream>>>(xFlipB, Wi1bP, b1bP, XpB, nullptr, (int)MT, G4_, D_, 0);
  }
  run_dir(XpF, Wh1fP, hsF);
  run_dir(XpB, Wh1bP, hsB);
  concat_unflip<<<gb(MT * (H2_ / 8)), 256, 0, stream>>>(hsF, hsB, lengths, x1B, T_);

  // ---- phase 3: BiLSTM layer 2 ----
  flip_bf16<<<gb(MT * (H2_ / 8)), 256, 0, stream>>>(x1B, xFlipB, lengths, T_, H2_);
  {
    dim3 g(G4_ / 128, (unsigned)(MT / 128));
    gemm_bf16_wmma<<<g, 256, 0, stream>>>(x1B,    Wi2fP, b2fP, XpF, nullptr, (int)MT, G4_, H2_, 0);
    gemm_bf16_wmma<<<g, 256, 0, stream>>>(xFlipB, Wi2bP, b2bP, XpB, nullptr, (int)MT, G4_, H2_, 0);
  }
  run_dir(XpF, Wh2fP, hsF);
  run_dir(XpB, Wh2bP, hsB);
  concat_unflip<<<gb(MT * (H2_ / 8)), 256, 0, stream>>>(hsF, hsB, lengths, x2B, T_);

  // ---- phase 4: attention + heads ----
  {
    dim3 g(H2_ / 128, (unsigned)(MT / 128));
    gemm_bf16_wmma<<<g, 256, 0, stream>>>(x2B, WkT, nullptr, keys, nullptr, (int)MT, H2_, H2_, 0);
  }
  scores_kernel<<<(unsigned)MT, 256, 0, stream>>>(keys, w_energy, lengths, scores, T_);
  softmax_kernel<<<B_, 256, 0, stream>>>(scores, attn, T_);
  context_kernel<<<dim3(B_, H2_ / 256), 256, 0, stream>>>(attn, x2B, ctxB, T_);
  {
    dim3 g(H2_ / 128, 1);   // M=32 tail handled by zero-pad/guard
    gemm_bf16_wmma<<<g, 256, 0, stream>>>(ctxB, WpT, b_post, ctxP, nullptr, B_, H2_, H2_, 0);
  }
  intents_kernel<<<gb(B_ * NI_), 256, 0, stream>>>(ctxP, W_int, b_int, out);
}
